// SimpleGCN_20212116095606
// MI455X (gfx1250) — matrix operand order repaired and checked by reading, not compiled, not run on hardware
//
#include <hip/hip_runtime.h>
#include <hip/hip_bf16.h>

typedef __attribute__((ext_vector_type(16))) __bf16 v16bf;
typedef __attribute__((ext_vector_type(8)))  float  v8f;

#define HID 256

__device__ __forceinline__ void atomAddF(float* p, float v) {
    unsafeAtomicAdd(p, v);   // non-returning global_atomic_add_f32
}

__device__ __forceinline__ void pack4(v16bf& v, int base, float4 f) {
    v[base + 0] = (__bf16)f.x; v[base + 1] = (__bf16)f.y;
    v[base + 2] = (__bf16)f.z; v[base + 3] = (__bf16)f.w;
}

// ---------------- degree / norm ----------------
__global__ void k_init_deg(float* deg, int n) {
    int i = blockIdx.x * blockDim.x + threadIdx.x;
    if (i < n) deg[i] = 1.0f;                      // self-loop
}

__global__ void k_accum_deg(const long long* __restrict__ ei, int E, float* deg) {
    int e = blockIdx.x * blockDim.x + threadIdx.x;
    if (e < E) atomAddF(&deg[(int)ei[(long)E + e]], 1.0f);   // dst side
}

__global__ void k_rsqrt_deg(float* deg, int n) {
    int i = blockIdx.x * blockDim.x + threadIdx.x;
    if (i < n) deg[i] = rsqrtf(deg[i]);            // deg >= 1 always
}

// ---------------- W (f32 [K x 256]) -> bf16 pre-swizzled into B-fragment order ----------
// Layout: Wb[(((ct*(K/32) + ks)*32) + lane)*16 + e]
//   lane = kh*16 + m ; col = ct*16 + m ; K-index = ks*32 + (e<8 ? kh*8 + e : 16 + kh*8 + e-8)
__global__ void k_convW(const float* __restrict__ W, __bf16* __restrict__ Wb, int K) {
    int idx = blockIdx.x * blockDim.x + threadIdx.x;   // over K*256
    if (idx >= K * HID) return;
    int k = idx >> 8;          // 0..K-1
    int c = idx & 255;         // 0..255
    int ct = c >> 4, m = c & 15;
    int ks = k >> 5, kh = (k >> 3) & 1, e = (k & 7) | ((k >> 1) & 8);
    int lane = kh * 16 + m;
    Wb[(((ct * (K >> 5) + ks) << 5) + lane) * 16 + e] = (__bf16)W[k * HID + c];
}

// ---------------- GEMM: H[N x 256] = A[N x K] @ W[K x 256] (bf16 WMMA, f32 acc) ------
// One block = 16 rows; 8 waves; each wave owns 2 adjacent col tiles (reuses A fragment).
template <int K>
__global__ void __launch_bounds__(256) k_gemm_bf16(
    const float* __restrict__ A, const __bf16* __restrict__ Wb,
    float* __restrict__ H) {
    const int lane = threadIdx.x & 31;
    const int wave = threadIdx.x >> 5;
    const int m    = lane & 15;       // row within tile / col within tile
    const int kh   = lane >> 4;       // which K-half this lane holds
    const long row = (long)blockIdx.x * 16 + m;
    const float* arow = A + row * (long)K;

    const int ct0 = wave * 2;
    const v16bf* bp0 = (const v16bf*)Wb + ((long)(ct0 + 0) * (K >> 5)) * 32 + lane;
    const v16bf* bp1 = (const v16bf*)Wb + ((long)(ct0 + 1) * (K >> 5)) * 32 + lane;

    v8f acc0 = {}, acc1 = {};
    #pragma unroll 4
    for (int ks = 0; ks < (K >> 5); ++ks) {
        // A fragment 16x32 bf16: lanes 0-15 hold K {0..7,16..23}, lanes 16-31 {8..15,24..31}
        const float4* ap = (const float4*)(arow + ks * 32 + kh * 8);
        float4 a0 = ap[0], a1 = ap[1], a2 = ap[4], a3 = ap[5];
        v16bf a;
        pack4(a, 0, a0);  pack4(a, 4, a1);
        pack4(a, 8, a2);  pack4(a, 12, a3);
        // B fragments: pre-swizzled, 32 contiguous bytes per lane
        v16bf b0 = bp0[ks * 32];
        v16bf b1 = bp1[ks * 32];
        acc0 = __builtin_amdgcn_wmma_f32_16x16x32_bf16(
            false, a, false, b0, (short)0, acc0, false, false);
        acc1 = __builtin_amdgcn_wmma_f32_16x16x32_bf16(
            false, a, false, b1, (short)0, acc1, false, false);
    }
    // C/D layout: lanes 0-15 -> M = v, lanes 16-31 -> M = 8+v, N = lane&15
    float* out0 = H + ((long)blockIdx.x * 16 + kh * 8) * HID + ct0 * 16 + m;
    #pragma unroll
    for (int v = 0; v < 8; ++v) out0[v * HID] = acc0[v];
    float* out1 = out0 + 16;
    #pragma unroll
    for (int v = 0; v < 8; ++v) out1[v * HID] = acc1[v];
}

// ---------------- agg = bias + dinv^2 * h  (self-loop + bias fused) ----------------
__global__ void k_init_agg(const float4* __restrict__ h4, const float* __restrict__ dinv,
                           const float* __restrict__ bias, float4* __restrict__ agg4, int n) {
    int idx = blockIdx.x * blockDim.x + threadIdx.x;   // over n*64 float4s
    if (idx >= n * 64) return;
    int node = idx >> 6, q = idx & 63;
    float di = dinv[node];
    float w  = di * di;
    float4 hv = h4[idx];
    float4 bv = ((const float4*)bias)[q];
    float4 r;
    r.x = bv.x + w * hv.x; r.y = bv.y + w * hv.y;
    r.z = bv.z + w * hv.z; r.w = bv.w + w * hv.w;
    agg4[idx] = r;
}

// ------- edge gather + scatter-add: one wave per edge, 8 ch/lane, next-edge prefetch ----
__global__ void __launch_bounds__(256) k_scatter(
    const long long* __restrict__ ei, int E,
    const float* __restrict__ dinv,
    const float* __restrict__ h,
    float* agg) {
    const int lane = threadIdx.x & 31;
    int wid = (blockIdx.x * blockDim.x + threadIdx.x) >> 5;
    const int nw = (gridDim.x * blockDim.x) >> 5;

    int e = wid;
    if (e >= E) return;
    int s = (int)ei[e];
    int d = (int)ei[(long)E + e];
    for (; e < E; ) {
        const int en = e + nw;
        int sn = 0, dn = 0;
        if (en < E) {
            sn = (int)ei[en];
            dn = (int)ei[(long)E + en];
            __builtin_prefetch(h + (long)sn * HID + lane * 8, 0, 0);  // warm next gather row
        }
        const float w = dinv[s] * dinv[d];
        const float4* hp = (const float4*)(h + (long)s * HID) + lane * 2;
        float4 m0 = hp[0], m1 = hp[1];
        float* op = agg + (long)d * HID + lane * 8;
        atomAddF(op + 0, w * m0.x); atomAddF(op + 1, w * m0.y);
        atomAddF(op + 2, w * m0.z); atomAddF(op + 3, w * m0.w);
        atomAddF(op + 4, w * m1.x); atomAddF(op + 5, w * m1.y);
        atomAddF(op + 6, w * m1.z); atomAddF(op + 7, w * m1.w);
        s = sn; d = dn; e = en;
    }
}

// ---------------- in-place ReLU ----------------
__global__ void k_relu(float4* buf, int n4) {
    int i = blockIdx.x * blockDim.x + threadIdx.x;
    if (i >= n4) return;
    float4 v = buf[i];
    v.x = fmaxf(v.x, 0.0f); v.y = fmaxf(v.y, 0.0f);
    v.z = fmaxf(v.z, 0.0f); v.w = fmaxf(v.w, 0.0f);
    buf[i] = v;
}

// ---------------- readout: out[i] = h[i,:] . Wout + bout ----------------
__global__ void __launch_bounds__(256) k_readout(
    const float* __restrict__ h, const float* __restrict__ Wout,
    const float* __restrict__ bout, float* __restrict__ out, int n) {
    int lane = threadIdx.x & 31, wave = threadIdx.x >> 5;
    int node = blockIdx.x * 8 + wave;
    if (node >= n) return;
    const float4* hp = (const float4*)(h + (long)node * HID) + lane * 2;
    const float4* wp = (const float4*)Wout + lane * 2;
    float4 a0 = hp[0], a1 = hp[1], w0 = wp[0], w1 = wp[1];
    float s = a0.x * w0.x + a0.y * w0.y + a0.z * w0.z + a0.w * w0.w
            + a1.x * w1.x + a1.y * w1.y + a1.z * w1.z + a1.w * w1.w;
    #pragma unroll
    for (int o = 16; o > 0; o >>= 1) s += __shfl_xor(s, o, 32);
    if (lane == 0) out[node] = s + bout[0];
}

extern "C" void kernel_launch(void* const* d_in, const int* in_sizes, int n_in,
                              void* d_out, int out_size, void* d_ws, size_t ws_size,
                              hipStream_t stream) {
    const float*     x  = (const float*)d_in[0];
    const long long* ei = (const long long*)d_in[1];
    const float* W0 = (const float*)d_in[2];
    const float* b0 = (const float*)d_in[3];
    const float* W1 = (const float*)d_in[4];
    const float* b1 = (const float*)d_in[5];
    const float* W2 = (const float*)d_in[6];
    const float* b2 = (const float*)d_in[7];
    const float* Wo = (const float*)d_in[8];
    const float* bo = (const float*)d_in[9];
    float* out = (float*)d_out;

    const int IN = 512;
    const int N  = in_sizes[0] / IN;     // 50000 (multiple of 16)
    const int E  = in_sizes[1] / 2;      // 800000

    // workspace: dinv [N] | bufA [N*256] f32 | bufB [N*256] f32 | Wb [512*256] bf16
    float* dinv = (float*)d_ws;
    size_t off  = ((size_t)N * sizeof(float) + 255) & ~(size_t)255;
    float* bufA = (float*)((char*)d_ws + off);
    float* bufB = bufA + (size_t)N * HID;
    __bf16* Wb  = (__bf16*)(bufB + (size_t)N * HID);

    dim3 blk(256);
    const int gN    = (N + 255) / 256;
    const int gE    = (E + 255) / 256;
    const int gGemm = N / 16;
    const int n4    = N * 64;
    const int g4    = (n4 + 255) / 256;
    const int gW512 = (512 * HID + 255) / 256;
    const int gW256 = (256 * HID + 255) / 256;

    // degree / normalization
    k_init_deg <<<gN, blk, 0, stream>>>(dinv, N);
    k_accum_deg<<<gE, blk, 0, stream>>>(ei, E, dinv);
    k_rsqrt_deg<<<gN, blk, 0, stream>>>(dinv, N);

    // layer 1: x[512] -> bufA -> bufB
    k_convW<<<gW512, blk, 0, stream>>>(W0, Wb, 512);
    k_gemm_bf16<512><<<gGemm, blk, 0, stream>>>(x, Wb, bufA);
    k_init_agg <<<g4, blk, 0, stream>>>((const float4*)bufA, dinv, b0, (float4*)bufB, N);
    k_scatter  <<<2048, blk, 0, stream>>>(ei, E, dinv, bufA, bufB);
    k_relu     <<<g4, blk, 0, stream>>>((float4*)bufB, n4);

    // layer 2: bufB[256] -> bufA -> bufB
    k_convW<<<gW256, blk, 0, stream>>>(W1, Wb, 256);
    k_gemm_bf16<256><<<gGemm, blk, 0, stream>>>(bufB, Wb, bufA);
    k_init_agg <<<g4, blk, 0, stream>>>((const float4*)bufA, dinv, b1, (float4*)bufB, N);
    k_scatter  <<<2048, blk, 0, stream>>>(ei, E, dinv, bufA, bufB);
    k_relu     <<<g4, blk, 0, stream>>>((float4*)bufB, n4);

    // layer 3
    k_convW<<<gW256, blk, 0, stream>>>(W2, Wb, 256);
    k_gemm_bf16<256><<<gGemm, blk, 0, stream>>>(bufB, Wb, bufA);
    k_init_agg <<<g4, blk, 0, stream>>>((const float4*)bufA, dinv, b2, (float4*)bufB, N);
    k_scatter  <<<2048, blk, 0, stream>>>(ei, E, dinv, bufA, bufB);
    k_relu     <<<g4, blk, 0, stream>>>((float4*)bufB, n4);

    // readout
    k_readout<<<(N + 7) / 8, blk, 0, stream>>>(bufB, Wo, bo, out, N);
}
